// ScaledDotProductAttention_9131100471600
// MI455X (gfx1250) — matrix operand order repaired and checked by reading, not compiled, not run on hardware
//
#include <hip/hip_runtime.h>

// MI455X / gfx1250 fused scaled-dot-product attention (forward, returns
// output + full attention weights). wave32, v_wmma_f32_16x16x32_f16.
//
// Per workgroup (128 thr = 4 waves): 16 query rows.
//   Phase A: scores = scale*Q*K^T + mask -> LDS (16x2048 fp32, 128 KB),
//            K staged in 128-key fp32 chunks via coalesced b128 loads.
//   Phase B: row softmax in LDS, normalized weights streamed to HBM.
//   Phase C: O = P*V, V staged in 128-key fp32 chunks in LDS.
// LDS: 128 KB scores + 32 KB stage = 160 KB -> 2 workgroups per 320 KB WGP.

typedef __attribute__((ext_vector_type(16))) _Float16 v16h;
typedef __attribute__((ext_vector_type(8)))  float    v8f;
typedef __attribute__((ext_vector_type(4)))  float    vf4;

#define B_    2
#define H_    16
#define S_    2048
#define D_    64
#define QT    16            // query rows per workgroup
#define CK    128           // keys per staged chunk
#define NCH   (S_ / CK)     // 16 chunks
#define SOFF  (QT * S_)     // float offset of staging region in LDS

__global__ __launch_bounds__(128)
void attn_fwd_gfx1250(const float* __restrict__ Q, const float* __restrict__ K,
                      const float* __restrict__ V, const int* __restrict__ mask,
                      float* __restrict__ out, float* __restrict__ attnw)
{
  extern __shared__ float smem[];
  float* sS = smem;            // [QT][S_]  score strip (fp32)
  float* sT = smem + SOFF;     // [CK][D_]  K/V staging chunk (fp32)

  const int tid  = threadIdx.x;
  const int lane = tid & 31;
  const int wave = tid >> 5;
  const int half = lane >> 4;   // 0: lanes 0-15, 1: lanes 16-31
  const int l16  = lane & 15;

  const int qblk = blockIdx.x;          // 0..S/16-1
  const int bh   = blockIdx.y;          // 0..B*H-1
  const int b    = bh / H_;
  const int q0   = qblk * QT;

  const float scale = 0.125f;           // 1/sqrt(64)

  const float* Qb = Q + (size_t)bh * S_ * D_;
  const float* Kb = K + (size_t)bh * S_ * D_;
  const float* Vb = V + (size_t)bh * S_ * D_;
  const int*   Mb = mask + (size_t)b * S_ * S_;

  // ---- Q fragment: A-matrix 16x32 f16, two d-chunks (0..31, 32..63) ----
  // ISA 7.12.2 16-bit A 16x32: lane row M = lane%16; element e of the
  // fragment holds K = (e<8 ? 0 : 16) + (lane/16)*8 + (e%8).
  v16h a0, a1;
  {
    const float* Qrow = Qb + (size_t)(q0 + l16) * D_;
    #pragma unroll
    for (int e = 0; e < 8; ++e) {
      a0[e]     = (_Float16)Qrow[half * 8 + e];
      a0[8 + e] = (_Float16)Qrow[16 + half * 8 + e];
      a1[e]     = (_Float16)Qrow[32 + half * 8 + e];
      a1[8 + e] = (_Float16)Qrow[48 + half * 8 + e];
    }
  }

  // ---- Phase A: scores = scale * Q K^T, mask, -> LDS ----
  for (int c = 0; c < NCH; ++c) {
    const int kc0 = c * CK;
    __syncthreads();                       // stage buffer safe to overwrite
    {
      // coalesced stage: thread t copies K row (kc0+t), 64 floats = 16x b128
      const vf4* src = (const vf4*)(Kb + (size_t)(kc0 + tid) * D_);
      vf4*       dst = (vf4*)(sT + tid * D_);
      #pragma unroll
      for (int j = 0; j < D_ / 4; ++j) dst[j] = src[j];
      if (c + 1 < NCH)                     // global_prefetch_b8 of next chunk
        __builtin_prefetch(Kb + (size_t)(kc0 + CK + tid) * D_, 0, 0);
    }
    __syncthreads();

    #pragma unroll
    for (int t = 0; t < 2; ++t) {          // 8 key tiles / chunk, 2 per wave
      const int tloc = wave * 2 + t;
      const int key0 = kc0 + tloc * 16;
      // B-matrix 32x16 f16: lane col N = lane%16; element e holds
      // K = (e<8 ? 0 : 16) + (lane/16)*8 + (e%8). B[k][n] = K[key0+n][d=k].
      const float* kr = sT + (tloc * 16 + l16) * D_;
      v16h b0, b1;
      #pragma unroll
      for (int e = 0; e < 8; ++e) {
        b0[e]     = (_Float16)kr[half * 8 + e];
        b0[8 + e] = (_Float16)kr[16 + half * 8 + e];
        b1[e]     = (_Float16)kr[32 + half * 8 + e];
        b1[8 + e] = (_Float16)kr[48 + half * 8 + e];
      }
      v8f acc = {};
      acc = __builtin_amdgcn_wmma_f32_16x16x32_f16(false, a0, false, b0,
                                                   (short)0, acc, false, false);
      acc = __builtin_amdgcn_wmma_f32_16x16x32_f16(false, a1, false, b1,
                                                   (short)0, acc, false, false);
      // C layout: VGPR r -> row M = r + 8*(lane/16), col N = lane%16
      #pragma unroll
      for (int r = 0; r < 8; ++r) {
        int m    = r + 8 * half;
        int kcol = key0 + l16;
        float s  = acc[r] * scale;
        if (Mb[(size_t)(q0 + m) * S_ + kcol] == 0) s = -1e9f;
        sS[m * S_ + kcol] = s;
      }
    }
  }
  __syncthreads();

  // ---- Phase B: row softmax in LDS; stream normalized weights to HBM ----
  for (int r = wave * 4; r < wave * 4 + 4; ++r) {
    float* row = sS + r * S_;
    float mx = -3.0e38f;
    for (int cc = lane; cc < S_; cc += 32) mx = fmaxf(mx, row[cc]);
    #pragma unroll
    for (int off = 16; off > 0; off >>= 1)
      mx = fmaxf(mx, __shfl_xor(mx, off, 32));

    float sum = 0.f;
    for (int cc = lane; cc < S_; cc += 32) {
      float e = __expf(row[cc] - mx);
      row[cc] = e;
      sum += e;
    }
    #pragma unroll
    for (int off = 16; off > 0; off >>= 1)
      sum += __shfl_xor(sum, off, 32);

    float inv = 1.0f / sum;
    float* gW = attnw + ((size_t)bh * S_ + (size_t)(q0 + r)) * S_;
    for (int cc = lane; cc < S_; cc += 32) {   // coalesced 128B stores
      float p = row[cc] * inv;
      row[cc] = p;
      gW[cc]  = p;
    }
  }

  // ---- Phase C: O = P * V (each wave owns a 16-wide d-slice) ----
  const int dc = wave * 16;
  v8f o = {};
  for (int c = 0; c < NCH; ++c) {
    const int kc0 = c * CK;
    __syncthreads();                       // stage buffer safe to overwrite
    {
      const vf4* src = (const vf4*)(Vb + (size_t)(kc0 + tid) * D_);
      vf4*       dst = (vf4*)(sT + tid * D_);
      #pragma unroll
      for (int j = 0; j < D_ / 4; ++j) dst[j] = src[j];
      if (c + 1 < NCH)
        __builtin_prefetch(Vb + (size_t)(kc0 + CK + tid) * D_, 0, 0);
    }
    __syncthreads();

    #pragma unroll
    for (int kk = 0; kk < CK; kk += 32) {
      // A fragment from LDS scores: two contiguous 8-float runs per lane
      v16h pa;
      const float* prow = sS + l16 * S_ + kc0 + kk;
      #pragma unroll
      for (int e = 0; e < 8; ++e) {
        pa[e]     = (_Float16)prow[half * 8 + e];
        pa[8 + e] = (_Float16)prow[16 + half * 8 + e];
      }
      // B fragment from staged V: B[k][n] = V[kc0+kk+k][dc+n];
      // element e holds key (e<8 ? 0 : 16) + (lane/16)*8 + (e%8), fixed d = dc + lane%16.
      v16h vb;
      const float* vcol = sT + (kk + half * 8) * D_ + dc + l16;
      #pragma unroll
      for (int e = 0; e < 8; ++e) {
        vb[e]     = (_Float16)vcol[e * D_];    // ds_load_b32, conflict-free
        vb[8 + e] = (_Float16)vcol[(16 + e) * D_];
      }
      o = __builtin_amdgcn_wmma_f32_16x16x32_f16(false, pa, false, vb,
                                                 (short)0, o, false, false);
    }
  }
  #pragma unroll
  for (int r = 0; r < 8; ++r) {
    int m = r + 8 * half;
    out[((size_t)bh * S_ + (size_t)(q0 + m)) * D_ + dc + l16] = o[r];
  }
}

extern "C" void kernel_launch(void* const* d_in, const int* in_sizes, int n_in,
                              void* d_out, int out_size, void* d_ws, size_t ws_size,
                              hipStream_t stream) {
  const float* Q    = (const float*)d_in[0];
  const float* K    = (const float*)d_in[1];
  const float* V    = (const float*)d_in[2];
  const int*   mask = (const int*)d_in[3];

  float* out   = (float*)d_out;                        // [B,H,S,D]
  float* attnw = out + (size_t)B_ * H_ * S_ * D_;      // [B,H,S,S]

  dim3 grid(S_ / QT, B_ * H_);
  dim3 block(128);
  size_t smem = (size_t)(QT * S_ + CK * D_) * sizeof(float);   // 160 KB LDS
  attn_fwd_gfx1250<<<grid, block, smem, stream>>>(Q, K, V, mask, out, attnw);
}
